// GCNEncoder_68023692034099
// MI455X (gfx1250) — matrix operand order repaired and checked by reading, not compiled, not run on hardware
//
#include <hip/hip_runtime.h>
#include <cstdint>
#include <cstddef>

// ---------------------------------------------------------------------------
// CDNA5 (gfx1250) GCN forward: 3x (bf16-WMMA GEMM -> init(self-loop+bias) ->
// per-edge gather/atomic-scatter -> ReLU). wave32 everywhere.
// ---------------------------------------------------------------------------

typedef __bf16 v16bf __attribute__((ext_vector_type(16)));
typedef float  v8f   __attribute__((ext_vector_type(8)));

// Exact RNE (off hot path: weight swizzle only)
__device__ __forceinline__ uint32_t bfbits(float f) {
  union { float f; uint32_t u; } v; v.f = f;
  return (v.u + 0x7FFFu + ((v.u >> 16) & 1u)) >> 16;
}

// Hot path: round-to-nearest (ties away) + single V_PERM_B32 pack.
// result = { bf16(hi), bf16(lo) }  -> 3 VALU ops per packed pair.
__device__ __forceinline__ uint32_t pk_bf16(float lo, float hi) {
  union { float f; uint32_t u; } a, b;
  a.f = lo; b.f = hi;
  return __builtin_amdgcn_perm(b.u + 0x8000u, a.u + 0x8000u, 0x07060302u);
}

// ---------------------------------------------------------------------------
// Pre-swizzle W[128, OUT] (fp32) into wave32 B-fragment order (bf16):
//   element index = (((ct*4 + kk)*32 + lane)*16 + e)
//   maps matrix element (k = kk*32 + 8*(lane>>4) + (e&7) + 16*(e>>3), n = ct*16 + (lane&15))
// Each (ct,kk,lane) owns 32 contiguous bytes -> GEMM loads one v16bf per lane.
// ---------------------------------------------------------------------------
template<int OUT>
__global__ void wswizzle_k(const float* __restrict__ W, __bf16* __restrict__ Wsw) {
  int i = blockIdx.x * blockDim.x + threadIdx.x;     // total = OUT*128
  if (i >= OUT * 128) return;
  const int e    = i & 15;
  const int lane = (i >> 4) & 31;
  const int kk   = (i >> 9) & 3;
  const int ct   = i >> 11;
  const int k = kk * 32 + (lane >> 4) * 8 + (e & 7) + (e >> 3) * 16;
  const int n = ct * 16 + (lane & 15);
  union { uint16_t s; __bf16 b; } o;
  o.s = (uint16_t)bfbits(W[(size_t)k * OUT + n]);
  Wsw[i] = o.b;
}

// ---------------------------------------------------------------------------
// GEMM: H[N, OUT] = X[N, 128] @ W[128, OUT]   (fp32 in/out, bf16 WMMA core)
// Block = 128 threads = 4 waves; each wave owns OUT/64 column tiles of a
// 16-row tile; K=128 swept in four 16x16x32 bf16 WMMA steps.
//
// A 16x32 bf16 layout (ISA 7.12.2): lane(half=lane>>4, m=lane&15);
//   dword pair p -> K = 2*(p&3) + 16*(p>>2) + 8*half
//   => per lane, two contiguous 8-float spans: [8*half, +8) and [16+8*half, +8)
// C/D 16x16 f32: vgpr r -> M = 8*half + r, N = lane&15
// ---------------------------------------------------------------------------
template<int OUT>
__global__ __launch_bounds__(128) void gemm_bf16_wmma(
    const float* __restrict__ X, const __bf16* __restrict__ Wsw,
    float* __restrict__ H, int nRows) {
  const int lane = threadIdx.x & 31;
  const int wave = threadIdx.x >> 5;
  const int half = lane >> 4;
  const int l16  = lane & 15;
  const int rowTile = blockIdx.x;
  constexpr int NT = OUT / 64;   // column tiles handled by each of 4 waves

  v8f acc[NT];
#pragma unroll
  for (int t = 0; t < NT; ++t)
#pragma unroll
    for (int r = 0; r < 8; ++r) acc[t][r] = 0.0f;

  int row = rowTile * 16 + l16;
  if (row >= nRows) row = nRows - 1;               // clamp: loads stay valid,
  const float* xrow = X + (size_t)row * 128;       // excess D rows not stored
  const v16bf* Wv = (const v16bf*)Wsw;

#pragma unroll
  for (int kk = 0; kk < 4; ++kk) {
    const int kbase = kk * 32;
    // ---- A fragment: 4 x global_load_b128 + V_PERM_B32 packing
    const float4 r0 = *(const float4*)(xrow + kbase + half * 8);
    const float4 r1 = *(const float4*)(xrow + kbase + half * 8 + 4);
    const float4 r2 = *(const float4*)(xrow + kbase + 16 + half * 8);
    const float4 r3 = *(const float4*)(xrow + kbase + 16 + half * 8 + 4);
    union { v16bf v; uint32_t u[8]; } a;
    a.u[0] = pk_bf16(r0.x, r0.y);  a.u[1] = pk_bf16(r0.z, r0.w);
    a.u[2] = pk_bf16(r1.x, r1.y);  a.u[3] = pk_bf16(r1.z, r1.w);
    a.u[4] = pk_bf16(r2.x, r2.y);  a.u[5] = pk_bf16(r2.z, r2.w);
    a.u[6] = pk_bf16(r3.x, r3.y);  a.u[7] = pk_bf16(r3.z, r3.w);
    // ---- B fragments (pre-swizzled, one 32B load) + WMMA per column tile
#pragma unroll
    for (int t = 0; t < NT; ++t) {
      const int ct = wave + t * 4;
      const v16bf bfrag = Wv[(size_t)(ct * 4 + kk) * 32 + lane];
      acc[t] = __builtin_amdgcn_wmma_f32_16x16x32_bf16(
          false, a.v, false, bfrag, (short)0, acc[t], false, false);
    }
  }

  // ---- store D tiles (no bias here: reference adds b after aggregation)
  const int rbase = rowTile * 16 + half * 8;
#pragma unroll
  for (int t = 0; t < NT; ++t) {
    const int col = (wave + t * 4) * 16 + l16;
    float* hp = H + (size_t)rbase * OUT + col;
    if (rbase + 8 <= nRows) {                       // full tile: no guards
#pragma unroll
      for (int r = 0; r < 8; ++r) hp[(size_t)r * OUT] = acc[t][r];
    } else {
#pragma unroll
      for (int r = 0; r < 8; ++r)
        if (rbase + r < nRows) hp[(size_t)r * OUT] = acc[t][r];
    }
  }
}

// ---------------------------------------------------------------------------
// degree / dinv
// ---------------------------------------------------------------------------
__global__ void deg_init_k(float* __restrict__ deg, int n) {
  int i = blockIdx.x * blockDim.x + threadIdx.x;
  if (i < n) deg[i] = 1.0f;                        // self-loop
}
__global__ void deg_accum_k(const int* __restrict__ dst, float* __restrict__ deg, int e) {
  int i = blockIdx.x * blockDim.x + threadIdx.x;
  if (i < e) unsafeAtomicAdd(&deg[dst[i]], 1.0f);  // global_atomic_add_f32
}
__global__ void deg_rsqrt_k(float* __restrict__ deg, int n) {
  int i = blockIdx.x * blockDim.x + threadIdx.x;
  if (i < n) deg[i] = rsqrtf(deg[i]);              // deg >= 1 always
}

// ---------------------------------------------------------------------------
// agg = H * dinv^2 + b   (self-loop contribution + bias, scatter target init)
// ---------------------------------------------------------------------------
template<int OUT>
__global__ void init_agg_k(const float* __restrict__ H, const float* __restrict__ dinv,
                           const float* __restrict__ b, float* __restrict__ agg, int total) {
  int i = blockIdx.x * blockDim.x + threadIdx.x;
  if (i >= total) return;
  const int node = i / OUT;          // OUT is a power of two -> shift
  const int c    = i & (OUT - 1);
  const float di = dinv[node];
  agg[i] = fmaf(H[i], di * di, b[c]);
}

// ---------------------------------------------------------------------------
// Edge scatter: one wave32 per edge; float4 gather of h[src], per-element
// global fp32 atomics into agg[dst]. L2-resident working set (51MB << 192MB).
// ---------------------------------------------------------------------------
template<int OUT>
__global__ __launch_bounds__(256) void scatter_k(
    const float* __restrict__ H, const int* __restrict__ src,
    const int* __restrict__ dst, const float* __restrict__ dinv,
    float* __restrict__ agg, int nEdges) {
  const int lane = threadIdx.x & 31;
  const int edge = blockIdx.x * 8 + (threadIdx.x >> 5);
  if (edge >= nEdges) return;
  // streaming edge arrays: speculative prefetch (global_prefetch_b8)
  __builtin_prefetch(&src[edge + 4096], 0, 0);
  __builtin_prefetch(&dst[edge + 4096], 0, 0);
  const int s = src[edge];
  const int d = dst[edge];
  const float nrm = dinv[s] * dinv[d];
  const float* hs = H + (size_t)s * OUT;
  float* ad = agg + (size_t)d * OUT;
  if constexpr (OUT == 128) {
    float4 v = *(const float4*)(hs + lane * 4);
    unsafeAtomicAdd(&ad[lane * 4 + 0], v.x * nrm);
    unsafeAtomicAdd(&ad[lane * 4 + 1], v.y * nrm);
    unsafeAtomicAdd(&ad[lane * 4 + 2], v.z * nrm);
    unsafeAtomicAdd(&ad[lane * 4 + 3], v.w * nrm);
  } else {
    float2 v = *(const float2*)(hs + lane * 2);
    unsafeAtomicAdd(&ad[lane * 2 + 0], v.x * nrm);
    unsafeAtomicAdd(&ad[lane * 2 + 1], v.y * nrm);
  }
}

__global__ void relu_k(float* __restrict__ a, int total) {
  int i = blockIdx.x * blockDim.x + threadIdx.x;
  if (i < total) a[i] = fmaxf(a[i], 0.0f);
}

// ---------------------------------------------------------------------------
// host side
// ---------------------------------------------------------------------------
extern "C" void kernel_launch(void* const* d_in, const int* in_sizes, int n_in,
                              void* d_out, int out_size, void* d_ws, size_t ws_size,
                              hipStream_t stream) {
  (void)n_in; (void)out_size; (void)ws_size;
  const float* x  = (const float*)d_in[0];
  const int*   ei = (const int*)d_in[1];
  const float* W1 = (const float*)d_in[2];
  const float* b1 = (const float*)d_in[3];
  const float* W2 = (const float*)d_in[4];
  const float* b2 = (const float*)d_in[5];
  const float* W3 = (const float*)d_in[6];
  const float* b3 = (const float*)d_in[7];

  const int N = in_sizes[0] / 128;
  const int E = in_sizes[1] / 2;
  const int* src = ei;       // edge_index[0]
  const int* dst = ei + E;   // edge_index[1]

  char* ws = (char*)d_ws;
  auto align256 = [](size_t v) { return (v + 255) & ~(size_t)255; };
  size_t off = 0;
  float*  dinv = (float*)(ws + off);  off = align256(off + (size_t)N * 4);
  float*  Hbuf = (float*)(ws + off);  off = align256(off + (size_t)N * 128 * 4);
  float*  Abuf = (float*)(ws + off);  off = align256(off + (size_t)N * 128 * 4);
  __bf16* Wsw  = (__bf16*)(ws + off); off = align256(off + (size_t)128 * 128 * 2);

  const int TPB = 256;
  const int nbN    = (N + TPB - 1) / TPB;
  const int nbE    = (E + TPB - 1) / TPB;
  const int nbEdge = (E + 7) / 8;          // 8 waves (edges) per 256-thr block
  const int nb128  = (int)(((size_t)N * 128 + TPB - 1) / TPB);
  const int nb64   = (int)(((size_t)N * 64 + TPB - 1) / TPB);
  const int nbW128 = (128 * 128 + TPB - 1) / TPB;
  const int nbW64  = (64 * 128 + TPB - 1) / TPB;
  const int rowTiles = (N + 15) / 16;

  // ---- degree -> dinv (shared by all layers)
  deg_init_k<<<nbN, TPB, 0, stream>>>(dinv, N);
  deg_accum_k<<<nbE, TPB, 0, stream>>>(dst, dinv, E);
  deg_rsqrt_k<<<nbN, TPB, 0, stream>>>(dinv, N);

  // ---- layer 1: 128 -> 128, ReLU
  wswizzle_k<128><<<nbW128, TPB, 0, stream>>>(W1, Wsw);
  gemm_bf16_wmma<128><<<rowTiles, 128, 0, stream>>>(x, Wsw, Hbuf, N);
  init_agg_k<128><<<nb128, TPB, 0, stream>>>(Hbuf, dinv, b1, Abuf, N * 128);
  scatter_k<128><<<nbEdge, 256, 0, stream>>>(Hbuf, src, dst, dinv, Abuf, E);
  relu_k<<<nb128, TPB, 0, stream>>>(Abuf, N * 128);

  // ---- layer 2: 128 -> 128, ReLU
  wswizzle_k<128><<<nbW128, TPB, 0, stream>>>(W2, Wsw);
  gemm_bf16_wmma<128><<<rowTiles, 128, 0, stream>>>(Abuf, Wsw, Hbuf, N);
  init_agg_k<128><<<nb128, TPB, 0, stream>>>(Hbuf, dinv, b2, Abuf, N * 128);
  scatter_k<128><<<nbEdge, 256, 0, stream>>>(Hbuf, src, dst, dinv, Abuf, E);
  relu_k<<<nb128, TPB, 0, stream>>>(Abuf, N * 128);

  // ---- layer 3: 128 -> 64, no activation, aggregate straight into d_out
  float* out = (float*)d_out;
  wswizzle_k<64><<<nbW64, TPB, 0, stream>>>(W3, Wsw);
  gemm_bf16_wmma<64><<<rowTiles, 128, 0, stream>>>(Abuf, Wsw, Hbuf, N);
  init_agg_k<64><<<nb64, TPB, 0, stream>>>(Hbuf, dinv, b3, out, N * 64);
  scatter_k<64><<<nbEdge, 256, 0, stream>>>(Hbuf, src, dst, dinv, out, E);
}
